// Attention_81956565942967
// MI455X (gfx1250) — matrix operand order repaired and checked by reading, not compiled, not run on hardware
//
#include <hip/hip_runtime.h>
#include <hip/hip_bf16.h>

// ---------------------------------------------------------------------------
// Attention for MI455X (gfx1250, wave32, WMMA).
// Pipeline: Q=x*Wq ; KV=ctx*Wkv ; flash-attn per (b,h) ; out=AO*Wout+b.
// All matmuls in bf16 WMMA (v_wmma_f32_16x16x32_bf16), f32 accumulate.
// Software-pipelined (fetch->compute->commit) with double-buffered LDS;
// K-tiles / bf16 A-tiles staged via GLOBAL_LOAD_ASYNC_TO_LDS when available.
// Workspace: qb 16MB + kvb 32MB + aob 16MB = 64MB bf16.
// ---------------------------------------------------------------------------

typedef __attribute__((ext_vector_type(16))) __bf16 v16bf;
typedef __attribute__((ext_vector_type(8)))  __bf16 v8bf;
typedef __attribute__((ext_vector_type(8)))  float  v8f;

#define LOG2E 1.44269504088896340736f

#if defined(__has_builtin)
#  if __has_builtin(__builtin_amdgcn_global_load_async_to_lds_b128)
#    define HAS_ASYNC_LDS 1
#  endif
#endif
#ifndef HAS_ASYNC_LDS
#  define HAS_ASYNC_LDS 0
#endif

#if defined(__has_builtin)
#  if __has_builtin(__builtin_amdgcn_s_wait_asynccnt)
#    define WAIT_ASYNCCNT(n) __builtin_amdgcn_s_wait_asynccnt(n)
#  endif
#endif
#ifndef WAIT_ASYNCCNT
#  define WAIT_ASYNCCNT(n) asm volatile("s_wait_asynccnt %0" :: "i"(n) : "memory")
#endif

#if HAS_ASYNC_LDS
// Builtin signature (from toolchain diagnostic): param0 is a NON-CONST
// pointer to int __attribute__((vector_size(16))) in the global (AS1)
// address space; param1 is the LDS (AS3) counterpart; then imm offset, cpol.
typedef int async_v4i __attribute__((vector_size(4 * sizeof(int))));
typedef __attribute__((address_space(1))) async_v4i* async_gptr;
typedef __attribute__((address_space(3))) async_v4i* async_lptr;

// global (AS1) -> LDS (AS3) asynchronous 16B copy, tracked by ASYNCcnt.
static __device__ __forceinline__ void async_copy_b128(const void* g, void* l) {
    __builtin_amdgcn_global_load_async_to_lds_b128(
        (async_gptr)(void*)(g), (async_lptr)(l), 0, 0);
}
#endif

static constexpr int BATCH = 8;
static constexpr int NSEQ  = 1024;
static constexpr int DIM_  = 1024;
static constexpr int HEADS = 16;
static constexpr int DHEAD = 64;
static constexpr int INNER = 1024;   // HEADS*DHEAD

static __device__ __forceinline__ unsigned int pack_bf16x2(float a, float b) {
    __bf16 ha = (__bf16)a, hb = (__bf16)b;
    unsigned short ba = __builtin_bit_cast(unsigned short, ha);
    unsigned short bb = __builtin_bit_cast(unsigned short, hb);
    return (unsigned int)ba | ((unsigned int)bb << 16);
}

// ---------------------------------------------------------------------------
// Generic tiled GEMM:  C[M,N] = A[M,K] * B[K,N] (+bias)
// A: fp32 or bf16 row-major (bf16 path async-staged to LDS when available)
// B: fp32 row-major (converted + transposed into LDS -> WMMA-B layout)
// Block tile 128x128, K-step 32, 8 waves (4 in M x 2 in N), wave tile 32x64.
// Double-buffered LDS; per K-step: fetch(next) -> 8 WMMAs -> commit(next).
// ---------------------------------------------------------------------------
template<bool A_F32, bool OUT_F32, bool BIAS>
__global__ __launch_bounds__(256)
void gemm_bf16_wmma(const void* __restrict__ Ap, const float* __restrict__ Bp,
                    void* __restrict__ Cp, const float* __restrict__ bias,
                    int M, int Nn, int K)
{
    constexpr int LDA = 40;   // 32 + 8 pad, keeps 16B alignment (80B rows)
    constexpr int LDB = 40;
    __shared__ __align__(16) __bf16 As[2][128 * LDA];   // [m][k]
    __shared__ __align__(16) __bf16 Bs[2][128 * LDB];   // [n][k]  (B transposed)

    const int tid  = threadIdx.x;
    const int wave = tid >> 5;
    const int lane = tid & 31;
    const int half = lane >> 4;
    const int lr   = lane & 15;
    const int wm   = wave >> 1;            // 0..3  -> 32 rows each
    const int wn   = wave & 1;             // 0..1  -> 64 cols each
    const int m0   = blockIdx.x * 128;
    const int n0   = blockIdx.y * 128;

    // ---- fetch/commit (software pipeline stages) ---------------------------
    float4 ra_f[4];     // A tile in flight (f32 path): 4 float4 chunks/thread
    uint4  ra_h[2];     // A tile in flight (bf16, no-async fallback)
    float4 rb[2][2];    // B tile in flight: 2 (k-pair, 4n) chunks/thread

    auto fetchA = [&](int k0) {
        if (A_F32) {
            const float* A = (const float*)Ap;
            #pragma unroll
            for (int t = 0; t < 4; ++t) {
                int c = tid + t * 256, row = c >> 3, kc = (c & 7) * 4;
                ra_f[t] = *(const float4*)(A + (size_t)(m0 + row) * K + k0 + kc);
            }
        } else {
            const __bf16* A = (const __bf16*)Ap;
            #pragma unroll
            for (int t = 0; t < 2; ++t) {
                int c = tid + t * 256, row = c >> 2, kc = (c & 3) * 8;
                ra_h[t] = *(const uint4*)(A + (size_t)(m0 + row) * K + k0 + kc);
            }
        }
    };
#if HAS_ASYNC_LDS
    auto asyncA = [&](int k0, int bufi) {    // bf16-A path: DMA straight to LDS
        const __bf16* A = (const __bf16*)Ap;
        #pragma unroll
        for (int t = 0; t < 2; ++t) {
            int c = tid + t * 256, row = c >> 2, kc = (c & 3) * 8;
            async_copy_b128(A + (size_t)(m0 + row) * K + k0 + kc,
                            &As[bufi][row * LDA + kc]);
        }
    };
#endif
    auto fetchB = [&](int k0) {
        #pragma unroll
        for (int t = 0; t < 2; ++t) {
            int c = tid + t * 256, kp = c >> 5, nc = (c & 31) * 4;
            rb[t][0] = *(const float4*)(Bp + (size_t)(k0 + 2 * kp + 0) * Nn + n0 + nc);
            rb[t][1] = *(const float4*)(Bp + (size_t)(k0 + 2 * kp + 1) * Nn + n0 + nc);
        }
    };
    auto commitA = [&](int bufi) {
        if (A_F32) {
            #pragma unroll
            for (int t = 0; t < 4; ++t) {
                int c = tid + t * 256, row = c >> 3, kc = (c & 7) * 4;
                uint2 p;
                p.x = pack_bf16x2(ra_f[t].x, ra_f[t].y);
                p.y = pack_bf16x2(ra_f[t].z, ra_f[t].w);
                *(uint2*)&As[bufi][row * LDA + kc] = p;     // one ds_store_b64
            }
        } else {
            #pragma unroll
            for (int t = 0; t < 2; ++t) {
                int c = tid + t * 256, row = c >> 2, kc = (c & 3) * 8;
                *(uint4*)&As[bufi][row * LDA + kc] = ra_h[t];
            }
        }
    };
    auto commitB = [&](int bufi) {           // pack k-pairs -> ds_store_b32
        #pragma unroll
        for (int t = 0; t < 2; ++t) {
            int c = tid + t * 256, kp = c >> 5, nc = (c & 31) * 4;
            *(unsigned int*)&Bs[bufi][(nc + 0) * LDB + 2 * kp] =
                pack_bf16x2(rb[t][0].x, rb[t][1].x);
            *(unsigned int*)&Bs[bufi][(nc + 1) * LDB + 2 * kp] =
                pack_bf16x2(rb[t][0].y, rb[t][1].y);
            *(unsigned int*)&Bs[bufi][(nc + 2) * LDB + 2 * kp] =
                pack_bf16x2(rb[t][0].z, rb[t][1].z);
            *(unsigned int*)&Bs[bufi][(nc + 3) * LDB + 2 * kp] =
                pack_bf16x2(rb[t][0].w, rb[t][1].w);
        }
    };

    v8f acc[2][4];
    #pragma unroll
    for (int i = 0; i < 2; ++i)
        #pragma unroll
        for (int j = 0; j < 4; ++j)
            #pragma unroll
            for (int e = 0; e < 8; ++e) acc[i][j][e] = 0.0f;

    // ---- prologue: tile 0 into LDS buffer 0 --------------------------------
    bool asyncA_used = false;
#if HAS_ASYNC_LDS
    if (!A_F32) { asyncA(0, 0); asyncA_used = true; }
#endif
    if (!asyncA_used) fetchA(0);
    fetchB(0);
#if HAS_ASYNC_LDS
    if (asyncA_used) WAIT_ASYNCCNT(0);
#endif
    if (!asyncA_used) commitA(0);
    commitB(0);

    int buf = 0;
    for (int k0 = 0; k0 < K; k0 += 32) {
#if HAS_ASYNC_LDS
        if (!A_F32) WAIT_ASYNCCNT(0);      // async A for 'buf' landed (own ops)
#endif
        __syncthreads();                   // all waves' tiles visible
        const bool nxt = (k0 + 32) < K;
        if (nxt) {
#if HAS_ASYNC_LDS
            if (!A_F32) asyncA(k0 + 32, buf ^ 1);   // DMA next A tile to LDS
            else        fetchA(k0 + 32);
#else
            fetchA(k0 + 32);
#endif
            fetchB(k0 + 32);               // global loads fly under WMMAs
        }
        if (k0 + 64 < K) {                 // warm caches for tile after next
            if (A_F32)
                __builtin_prefetch((const float*)Ap +
                    (size_t)(m0 + (tid >> 3)) * K + k0 + 64 + (tid & 7) * 4, 0, 3);
            else
                __builtin_prefetch((const __bf16*)Ap +
                    (size_t)(m0 + (tid >> 2)) * K + k0 + 64 + (tid & 3) * 8, 0, 3);
            __builtin_prefetch(Bp +
                (size_t)(k0 + 64 + (tid >> 5)) * Nn + n0 + (tid & 31) * 4, 0, 3);
        }

        // ---- fragments (ISA VGPR layouts) + 8 WMMAs ------------------------
        v16bf afrag[2];
        #pragma unroll
        for (int mt = 0; mt < 2; ++mt) {
            const __bf16* p = &As[buf][(wm * 32 + mt * 16 + lr) * LDA + half * 8];
            v8bf lo = *(const v8bf*)p;              // K = half*8 .. +7
            v8bf hi = *(const v8bf*)(p + 16);       // K = 16+half*8 .. +7
            #pragma unroll
            for (int e = 0; e < 8; ++e) { afrag[mt][e] = lo[e]; afrag[mt][8 + e] = hi[e]; }
        }
        v16bf bfrag[4];
        #pragma unroll
        for (int nt = 0; nt < 4; ++nt) {
            const __bf16* p = &Bs[buf][(wn * 64 + nt * 16 + lr) * LDB + half * 8];
            v8bf lo = *(const v8bf*)p;              // K = half*8 .. +7
            v8bf hi = *(const v8bf*)(p + 16);       // K = 16+half*8 .. +7
            #pragma unroll
            for (int e = 0; e < 8; ++e) { bfrag[nt][e] = lo[e]; bfrag[nt][8 + e] = hi[e]; }
        }
        #pragma unroll
        for (int mt = 0; mt < 2; ++mt)
            #pragma unroll
            for (int nt = 0; nt < 4; ++nt)
                acc[mt][nt] = __builtin_amdgcn_wmma_f32_16x16x32_bf16(
                    false, afrag[mt], false, bfrag[nt],
                    (short)0, acc[mt][nt], false, false);

        if (nxt) {
            if (!asyncA_used) commitA(buf ^ 1);    // stores land while others compute
            commitB(buf ^ 1);
        }
        buf ^= 1;
    }

    // ---- epilogue: C layout M = r + 8*half, N = lr -------------------------
    #pragma unroll
    for (int mt = 0; mt < 2; ++mt) {
        #pragma unroll
        for (int nt = 0; nt < 4; ++nt) {
            const int col = n0 + wn * 64 + nt * 16 + lr;
            const float bv = BIAS ? bias[col] : 0.0f;
            #pragma unroll
            for (int r = 0; r < 8; ++r) {
                const int row = m0 + wm * 32 + mt * 16 + r + 8 * half;
                const float v = acc[mt][nt][r] + bv;
                if (OUT_F32) ((float*)Cp)[(size_t)row * Nn + col] = v;
                else         ((__bf16*)Cp)[(size_t)row * Nn + col] = (__bf16)v;
            }
        }
    }
}

// ---------------------------------------------------------------------------
// Flash attention: one block = (b, h, 128 query rows); 8 waves x 16 rows.
// K tiles DMA'd to double-buffered LDS via GLOBAL_LOAD_ASYNC_TO_LDS (when
// available); V tiles register-pipelined (need transpose); online softmax;
// key-prefix mask from lengths[b] via uniform loop bound.
// ---------------------------------------------------------------------------
__global__ __launch_bounds__(256)
void flash_attn_wmma(const __bf16* __restrict__ qb, const __bf16* __restrict__ kvb,
                     const int* __restrict__ lengths, __bf16* __restrict__ ao)
{
    constexpr int LDK = 72;   // 64 + 8 pad  (144B rows, 16B aligned)
    constexpr int LDV = 40;   // 32 + 8 pad
    constexpr int LDP = 40;
    __shared__ __align__(16) __bf16 Ks[2][32 * LDK];  // [key][d] (= B layout, QK^T)
    __shared__ __align__(16) __bf16 Vt[64 * LDV];     // [d][key] (= B layout, P*V)
    __shared__ __align__(16) __bf16 Ps[8 * 16 * LDP]; // per-wave P tile [16][32]

    const int tid   = threadIdx.x;
    const int wave  = tid >> 5;
    const int lane  = tid & 31;
    const int half  = lane >> 4;
    const int lr    = lane & 15;
    const int b     = blockIdx.z;
    const int h     = blockIdx.y;
    const int qrow0 = blockIdx.x * 128 + wave * 16;
    const int len   = lengths[b];
    const float scale = 0.125f;                       // 64^-0.5
    const int jend  = ((len + 31) & ~31) < NSEQ ? ((len + 31) & ~31) : NSEQ;

    // Q fragments for d=0..31 and d=32..63, kept in registers for all j.
    v16bf qf[2];
    #pragma unroll
    for (int kd = 0; kd < 2; ++kd) {
        const __bf16* p = qb + (size_t)(b * NSEQ + qrow0 + lr) * INNER
                             + h * DHEAD + kd * 32 + half * 8;
        v8bf lo = *(const v8bf*)p;
        v8bf hi = *(const v8bf*)(p + 16);
        #pragma unroll
        for (int e = 0; e < 8; ++e) { qf[kd][e] = lo[e]; qf[kd][8 + e] = hi[e]; }
    }

    // Each thread owns one (key, 8d) chunk per 32-key tile.
    const int skey = tid >> 3;
    const int sdc  = (tid & 7) * 8;
    v8bf vfetch;
    auto fetchV = [&](int j0) {
        vfetch = *(const v8bf*)(kvb + (size_t)(b * NSEQ + j0 + skey) * (2 * INNER)
                                    + INNER + h * DHEAD + sdc);
    };
    auto commitV = [&]() {
        #pragma unroll
        for (int e = 0; e < 8; ++e) Vt[(sdc + e) * LDV + skey] = vfetch[e];
    };
#if HAS_ASYNC_LDS
    auto asyncK = [&](int j0, int bufi) {
        async_copy_b128(kvb + (size_t)(b * NSEQ + j0 + skey) * (2 * INNER)
                            + h * DHEAD + sdc,
                        &Ks[bufi][skey * LDK + sdc]);
    };
#else
    uint4 kfetch;
    auto fetchK = [&](int j0) {
        kfetch = *(const uint4*)(kvb + (size_t)(b * NSEQ + j0 + skey) * (2 * INNER)
                                     + h * DHEAD + sdc);
    };
    auto commitK = [&]() { *(uint4*)&Ks[0][skey * LDK + sdc] = kfetch; };
#endif

    float mrow[8], lrow[8];
    v8f oacc[4];
    #pragma unroll
    for (int r = 0; r < 8; ++r) { mrow[r] = -3.0e38f; lrow[r] = 0.0f; }
    #pragma unroll
    for (int dt = 0; dt < 4; ++dt)
        #pragma unroll
        for (int e = 0; e < 8; ++e) oacc[dt][e] = 0.0f;

    __bf16* Pw = &Ps[wave * 16 * LDP];

    // prologue (len >= 1, tile 0 always live)
#if HAS_ASYNC_LDS
    asyncK(0, 0);
#else
    fetchK(0);
#endif
    fetchV(0);

    int kbuf = 0;
    for (int j0 = 0; j0 < jend; j0 += 32) {
        const bool nxt = (j0 + 32) < jend;
#if HAS_ASYNC_LDS
        if (nxt) asyncK(j0 + 32, kbuf ^ 1);   // next K tile DMAs during compute
        commitV();
        if (nxt) { WAIT_ASYNCCNT(1); }        // in-order: oldest (current) done
        else     { WAIT_ASYNCCNT(0); }
#else
        commitK();
        commitV();
#endif
        __syncthreads();                      // everyone's K/V tile visible
        if (nxt) {
            fetchV(j0 + 32);                  // next V flies under compute
#if !HAS_ASYNC_LDS
            fetchK(j0 + 32);
#endif
        }

        // ---- S = Q * K^T : 2 key-subtiles, accumulate over d (2 K-steps) ---
        v8f s[2];
        #pragma unroll
        for (int nt = 0; nt < 2; ++nt) {
            #pragma unroll
            for (int e = 0; e < 8; ++e) s[nt][e] = 0.0f;
            #pragma unroll
            for (int kd = 0; kd < 2; ++kd) {
                const __bf16* p = &Ks[kbuf][(nt * 16 + lr) * LDK + kd * 32 + half * 8];
                v8bf lo = *(const v8bf*)p;
                v8bf hi = *(const v8bf*)(p + 16);
                v16bf kb;
                #pragma unroll
                for (int e = 0; e < 8; ++e) { kb[e] = lo[e]; kb[8 + e] = hi[e]; }
                s[nt] = __builtin_amdgcn_wmma_f32_16x16x32_bf16(
                    false, qf[kd], false, kb, (short)0, s[nt], false, false);
            }
        }

        // ---- online softmax (rows live on 16-lane groups) ------------------
        const bool valid0 = (j0 + lr) < len;
        const bool valid1 = (j0 + 16 + lr) < len;
        #pragma unroll
        for (int r = 0; r < 8; ++r) {
            float s0 = valid0 ? s[0][r] * scale : -3.0e38f;
            float s1 = valid1 ? s[1][r] * scale : -3.0e38f;
            float cm = fmaxf(s0, s1);
            #pragma unroll
            for (int m = 1; m < 16; m <<= 1) cm = fmaxf(cm, __shfl_xor(cm, m, 32));
            const float mnew  = fmaxf(mrow[r], cm);
            const float alpha = exp2f((mrow[r] - mnew) * LOG2E);
            const float p0 = valid0 ? exp2f((s0 - mnew) * LOG2E) : 0.0f;
            const float p1 = valid1 ? exp2f((s1 - mnew) * LOG2E) : 0.0f;
            float ps = p0 + p1;
            #pragma unroll
            for (int m = 1; m < 16; m <<= 1) ps += __shfl_xor(ps, m, 32);
            lrow[r] = lrow[r] * alpha + ps;
            mrow[r] = mnew;
            #pragma unroll
            for (int dt = 0; dt < 4; ++dt) oacc[dt][r] *= alpha;
            // C layout -> P tile (row M = r+8*half, key cols lr / 16+lr)
            Pw[(r + 8 * half) * LDP + lr]      = (__bf16)p0;
            Pw[(r + 8 * half) * LDP + 16 + lr] = (__bf16)p1;
        }
        // wave-private LDS round-trip (C-layout -> A-layout); lanes are in
        // lockstep within a wave, only the DS counter must drain.
        asm volatile("s_wait_dscnt 0" ::: "memory");

        // ---- P as WMMA-A fragment, O += P * V ------------------------------
        v16bf pf;
        {
            const __bf16* p = &Pw[lr * LDP + half * 8];
            v8bf lo = *(const v8bf*)p;
            v8bf hi = *(const v8bf*)(p + 16);
            #pragma unroll
            for (int e = 0; e < 8; ++e) { pf[e] = lo[e]; pf[8 + e] = hi[e]; }
        }
        #pragma unroll
        for (int dt = 0; dt < 4; ++dt) {
            const __bf16* p = &Vt[(dt * 16 + lr) * LDV + half * 8];
            v8bf lo = *(const v8bf*)p;
            v8bf hi = *(const v8bf*)(p + 16);
            v16bf vb;
            #pragma unroll
            for (int e = 0; e < 8; ++e) { vb[e] = lo[e]; vb[8 + e] = hi[e]; }
            oacc[dt] = __builtin_amdgcn_wmma_f32_16x16x32_bf16(
                false, pf, false, vb, (short)0, oacc[dt], false, false);
        }
        __syncthreads();   // readers done before next commit/async restages
#if HAS_ASYNC_LDS
        kbuf ^= 1;
#endif
    }

    // ---- normalize and write [b, n, h*64+d] as bf16 ------------------------
    #pragma unroll
    for (int r = 0; r < 8; ++r) {
        const float inv  = 1.0f / lrow[r];
        const int   grow = b * NSEQ + qrow0 + r + 8 * half;
        #pragma unroll
        for (int dt = 0; dt < 4; ++dt)
            ao[(size_t)grow * INNER + h * DHEAD + dt * 16 + lr] =
                (__bf16)(oacc[dt][r] * inv);
    }
}

// ---------------------------------------------------------------------------
extern "C" void kernel_launch(void* const* d_in, const int* in_sizes, int n_in,
                              void* d_out, int out_size, void* d_ws, size_t ws_size,
                              hipStream_t stream)
{
    const float* x    = (const float*)d_in[0];
    const float* ctx  = (const float*)d_in[1];
    const int*   lens = (const int*)d_in[2];
    const float* Wq   = (const float*)d_in[3];
    const float* Wkv  = (const float*)d_in[4];
    const float* Wout = (const float*)d_in[5];
    const float* bout = (const float*)d_in[6];
    float*       out  = (float*)d_out;

    const size_t ROWS = (size_t)BATCH * NSEQ;            // 8192
    __bf16* qb  = (__bf16*)d_ws;                         // [8192,1024] bf16
    __bf16* kvb = qb  + ROWS * INNER;                    // [8192,2048] bf16
    __bf16* aob = kvb + ROWS * (2 * INNER);              // [8192,1024] bf16

    const dim3 blk(256);
    // Q = x * Wq
    gemm_bf16_wmma<true, false, false>
        <<<dim3(ROWS / 128, INNER / 128), blk, 0, stream>>>(
            x, Wq, qb, nullptr, (int)ROWS, INNER, DIM_);
    // KV = ctx * Wkv
    gemm_bf16_wmma<true, false, false>
        <<<dim3(ROWS / 128, (2 * INNER) / 128), blk, 0, stream>>>(
            ctx, Wkv, kvb, nullptr, (int)ROWS, 2 * INNER, DIM_);
    // attention
    flash_attn_wmma<<<dim3(NSEQ / 128, HEADS, BATCH), blk, 0, stream>>>(
        qb, kvb, lens, aob);
    // out = AO * Wout + bout (fp32)
    gemm_bf16_wmma<false, true, true>
        <<<dim3(ROWS / 128, DIM_ / 128), blk, 0, stream>>>(
            aob, Wout, out, bout, (int)ROWS, DIM_, INNER);
}
